// VSDN_VAE_FILTER_32933809225759
// MI455X (gfx1250) — matrix operand order repaired and checked by reading, not compiled, not run on hardware
//
#include <hip/hip_runtime.h>
#include <hip/hip_bf16.h>

// ---------------- problem constants ----------------
#define T_STEPS 200
#define B_DIM   64
#define S_DIM   8
#define D_DIM   128
#define R_DIM   512
#define Z_DIM   256
#define H_DIM   1024
#define CCOV    16
#define CH_DIM  64
#define DT_C        0.05f
#define SQRT_DT_C   0.22360679774997896f
#define LOG2PI_C    1.8378770664093453f

typedef __attribute__((ext_vector_type(16))) __bf16 bf16x16;
typedef __attribute__((ext_vector_type(8)))  float  f32x8;

// ---------------------------------------------------------------------------
// Generic bf16-WMMA GEMM:  C[M,N] = act( Abf16 @ Wt^T + bias )
//   A[row,k] = maybe_relu( A0[row,k] (+ A1[row/S_DIM, k]) )  (fp32 -> bf16 in reg,
//   native v_cvt_pk_bf16_f32 path)
//   Wt is bf16, [N,K] row-major (pre-transposed weights), leading dim ldw.
//   ACT: 0 = identity, 2 = exp, 3 = 0.1*tanh   (compile-time)
// Block = 128 threads (4 waves); wave computes a 16(M) x 64(N) strip
// (4 accumulators sharing one A fragment). Grid = (N/64, M/64).
// ---------------------------------------------------------------------------
template<int ACT, bool HASA1, bool RELU>
__global__ __launch_bounds__(128)
void wmma_gemm_kernel(const float* __restrict__ A0, const float* __restrict__ A1,
                      const __bf16* __restrict__ Wt, int ldw,
                      const float* __restrict__ bias,
                      float* __restrict__ C,
                      int M, int N, int K, int lda)
{
  const int lane = threadIdx.x & 31;
  const int wave = threadIdx.x >> 5;
  const int m0 = blockIdx.y * 64 + wave * 16;
  if (m0 >= M) return;
  const int n0   = blockIdx.x * 64;
  const int l15  = lane & 15;
  const int hi   = (lane >> 4) & 1;
  const int arow = m0 + l15;
  const int aklo = hi ? 16 : 0;    // A: lanes 0-15 -> K 0..15; 16-31 -> K 16..31
  const int bklo = hi ? 16 : 0;    // B: lanes 0-15 -> K 0..15; 16-31 -> K 16..31

  f32x8 acc0 = {}; f32x8 acc1 = {}; f32x8 acc2 = {}; f32x8 acc3 = {};

  const float* arowp  = A0 + (size_t)arow * lda;
  const float* arowp1 = HASA1 ? (A1 + (size_t)(arow / S_DIM) * lda) : nullptr;

  for (int kk = 0; kk < K; kk += 32) {
    // ---- A fragment (fp32 b128 loads, fused add/relu, native cvt to bf16) ----
    bf16x16 a;
    {
      const float* p0 = arowp + kk + aklo;
      union { float4 f4[4]; float f[16]; } av;
      av.f4[0] = *(const float4*)(p0 + 0);
      av.f4[1] = *(const float4*)(p0 + 4);
      av.f4[2] = *(const float4*)(p0 + 8);
      av.f4[3] = *(const float4*)(p0 + 12);
      if (HASA1) {
        const float* p1 = arowp1 + kk + aklo;
        union { float4 f4[4]; float f[16]; } bv;
        bv.f4[0] = *(const float4*)(p1 + 0);
        bv.f4[1] = *(const float4*)(p1 + 4);
        bv.f4[2] = *(const float4*)(p1 + 8);
        bv.f4[3] = *(const float4*)(p1 + 12);
        #pragma unroll
        for (int i = 0; i < 16; ++i) av.f[i] += bv.f[i];
      }
      if (RELU) {
        #pragma unroll
        for (int i = 0; i < 16; ++i) av.f[i] = fmaxf(av.f[i], 0.0f);
      }
      #pragma unroll
      for (int i = 0; i < 16; ++i) a[i] = (__bf16)av.f[i];   // v_cvt_pk_bf16_f32
    }
    // ---- 4 B fragments (direct 32B vector loads) + 4 WMMAs ----
    #pragma unroll
    for (int t = 0; t < 4; ++t) {
      const int n = n0 + t * 16 + l15;
      bf16x16 b = *(const bf16x16*)(Wt + (size_t)n * ldw + kk + bklo);
      if (t == 0) acc0 = __builtin_amdgcn_wmma_f32_16x16x32_bf16(false, a, false, b, (short)0, acc0, false, false);
      if (t == 1) acc1 = __builtin_amdgcn_wmma_f32_16x16x32_bf16(false, a, false, b, (short)0, acc1, false, false);
      if (t == 2) acc2 = __builtin_amdgcn_wmma_f32_16x16x32_bf16(false, a, false, b, (short)0, acc2, false, false);
      if (t == 3) acc3 = __builtin_amdgcn_wmma_f32_16x16x32_bf16(false, a, false, b, (short)0, acc3, false, false);
    }
  }

  // ---- epilogue: C layout lanes 0-15 -> M rows 0..7, lanes 16-31 -> 8..15 ----
  const int mbase = m0 + (hi ? 8 : 0);
  #pragma unroll
  for (int t = 0; t < 4; ++t) {
    const int n = n0 + t * 16 + l15;
    const float bv = bias ? bias[n] : 0.0f;
    f32x8 acc = (t == 0) ? acc0 : (t == 1) ? acc1 : (t == 2) ? acc2 : acc3;
    #pragma unroll
    for (int r = 0; r < 8; ++r) {
      float v = acc[r] + bv;
      if (ACT == 2) v = expf(v);
      if (ACT == 3) v = 0.1f * tanhf(v);
      C[(size_t)(mbase + r) * N + n] = v;
    }
  }
}

// ---- one-time per call: W[K,N] fp32 -> Wt[N,K] bf16 ----
__global__ __launch_bounds__(256)
void transpose_bf16_kernel(const float* __restrict__ W, __bf16* __restrict__ Wt,
                           int K, int N)
{
  int i = blockIdx.x * 256 + threadIdx.x;
  if (i >= K * N) return;
  int n = i / K;
  int k = i - n * K;
  Wt[i] = (__bf16)W[(size_t)k * N + n];
}

__global__ void zero_accum_kernel(float* accum) {
  if (threadIdx.x < 2) accum[threadIdx.x] = 0.0f;
}

// ---- z0 = tanh(cov MLP), broadcast over S; one block per batch element ----
__global__ __launch_bounds__(64)
void z0_kernel(const float* __restrict__ cov, const float* __restrict__ w1,
               const float* __restrict__ b1, const float* __restrict__ w2,
               const float* __restrict__ b2, float* __restrict__ z)
{
  __shared__ float scov[CCOV];
  __shared__ float hid[CH_DIM];
  const int b = blockIdx.x;
  const int t = threadIdx.x;
  if (t < CCOV) scov[t] = cov[b * CCOV + t];
  __syncthreads();
  float h = b1[t];
  #pragma unroll
  for (int k = 0; k < CCOV; ++k) h += scov[k] * w1[k * CH_DIM + t];
  hid[t] = fmaxf(h, 0.0f);
  __syncthreads();
  for (int zz = t; zz < Z_DIM; zz += CH_DIM) {
    float v = b2[zz];
    for (int k = 0; k < CH_DIM; ++k) v += hid[k] * w2[k * Z_DIM + zz];
    v = tanhf(v);
    #pragma unroll
    for (int s = 0; s < S_DIM; ++s)
      z[((b * S_DIM + s) * Z_DIM) + zz] = v;
  }
}

// ---- Euler-Maruyama + min-max normalize + KLD accumulation.
//      One wave32 per (b,s) row (Z=256 -> 8 values/lane). ----
__global__ __launch_bounds__(256)
void update_kernel(const float* __restrict__ prior, const float* __restrict__ poster,
                   const float* __restrict__ sigma, const float* __restrict__ eps,
                   float* __restrict__ z, float* __restrict__ accum)
{
  const int lane = threadIdx.x & 31;
  const int wave = threadIdx.x >> 5;
  const int row  = blockIdx.x * 8 + wave;   // row = b*S + s
  const int b    = row >> 3;                // /S
  float zn[8];
  float mx = -3.0e38f, mn = 3.0e38f, es = 0.0f;
  #pragma unroll
  for (int j = 0; j < 8; ++j) {
    const int zi  = lane + j * 32;
    const int idx = row * Z_DIM + zi;
    const float sg = sigma[b * Z_DIM + zi];
    const float pd = poster[idx];
    const float v  = z[idx] + DT_C * pd + SQRT_DT_C * sg * eps[idx];
    zn[j] = v;
    mx = fmaxf(mx, v);
    mn = fminf(mn, v);
    const float e = (prior[idx] - pd) / sg;
    es += e * e;
  }
  #pragma unroll
  for (int o = 16; o > 0; o >>= 1) {
    mx = fmaxf(mx, __shfl_xor(mx, o, 32));
    mn = fminf(mn, __shfl_xor(mn, o, 32));
    es += __shfl_xor(es, o, 32);
  }
  const float inv = 1.0f / (mx - mn);
  #pragma unroll
  for (int j = 0; j < 8; ++j) {
    const int idx = row * Z_DIM + lane + j * 32;
    z[idx] = (zn[j] - mn) * inv;
  }
  if (lane == 0)
    atomicAdd(&accum[0], (0.5f * DT_C / (float)(B_DIM * S_DIM)) * es);
}

// ---- Gaussian NLL reconstruction term; one block per batch element ----
__global__ __launch_bounds__(128)
void recon_kernel(const float* __restrict__ Xt, const float* __restrict__ Mt,
                  const float* __restrict__ p, float* __restrict__ accum)
{
  __shared__ float sred[128];
  const int b = blockIdx.x;
  const int d = threadIdx.x;
  sred[d] = Mt[b * D_DIM + d];
  __syncthreads();
  for (int s = 64; s > 0; s >>= 1) {
    if (d < s) sred[d] += sred[d + s];
    __syncthreads();
  }
  const float Mtb = sred[0] * (1.0f / (float)D_DIM);
  __syncthreads();
  const float x = Xt[b * D_DIM + d];
  float acc = 0.0f;
  #pragma unroll
  for (int s = 0; s < S_DIM; ++s) {
    float diff = x - p[((b * S_DIM + s) * D_DIM) + d];
    acc += 0.5f * (LOG2PI_C + diff * diff);
  }
  sred[d] = acc;
  __syncthreads();
  for (int s = 64; s > 0; s >>= 1) {
    if (d < s) sred[d] += sred[d + s];
    __syncthreads();
  }
  if (d == 0)
    atomicAdd(&accum[1], Mtb * sred[0] * (1.0f / (float)(B_DIM * S_DIM)));
}

__global__ void finalize_kernel(const float* __restrict__ z,
                                const float* __restrict__ accum,
                                float* __restrict__ out)
{
  const int NZ = B_DIM * S_DIM * Z_DIM;
  int i = blockIdx.x * blockDim.x + threadIdx.x;
  if (i < NZ) out[i] = z[i];
  if (i == 0) {
    const float kld = accum[0], recon = accum[1];
    out[NZ + 0] = recon + 1.0f * kld;   // LAMBDA = 1.0
    out[NZ + 1] = recon;
    out[NZ + 2] = kld;
  }
}

// ---------------------------------------------------------------------------
extern "C" void kernel_launch(void* const* d_in, const int* in_sizes, int n_in,
                              void* d_out, int out_size, void* d_ws, size_t ws_size,
                              hipStream_t stream)
{
  (void)in_sizes; (void)n_in; (void)out_size; (void)ws_size;

  const float* X         = (const float*)d_in[0];
  const float* Mmask     = (const float*)d_in[1];
  const float* cov       = (const float*)d_in[2];
  const float* path_h    = (const float*)d_in[3];
  const float* path_hpos = (const float*)d_in[4];
  const float* noise     = (const float*)d_in[5];
  const float* drift_w1  = (const float*)d_in[6];
  const float* drift_b1  = (const float*)d_in[7];
  const float* drift_w2  = (const float*)d_in[8];
  const float* drift_b2  = (const float*)d_in[9];
  const float* diff_w1   = (const float*)d_in[10];
  const float* diff_b1   = (const float*)d_in[11];
  const float* diff_w2   = (const float*)d_in[12];
  const float* diff_b2   = (const float*)d_in[13];
  const float* p_w1      = (const float*)d_in[14];
  const float* p_b1      = (const float*)d_in[15];
  const float* p_w2      = (const float*)d_in[16];
  const float* p_b2      = (const float*)d_in[17];
  const float* cov_w1    = (const float*)d_in[18];
  const float* cov_b1    = (const float*)d_in[19];
  const float* cov_w2    = (const float*)d_in[20];
  const float* cov_b2    = (const float*)d_in[21];

  char* ws = (char*)d_ws;
  size_t off = 0;
  auto carve = [&](size_t bytes) -> void* {
    void* p = ws + off;
    off = (off + bytes + 255) & ~(size_t)255;
    return p;
  };

  // bf16 transposed weights [N,K]
  __bf16* wt_d1 = (__bf16*)carve((size_t)H_DIM * (R_DIM + Z_DIM) * 2);
  __bf16* wt_d2 = (__bf16*)carve((size_t)Z_DIM * H_DIM * 2);
  __bf16* wt_f1 = (__bf16*)carve((size_t)H_DIM * R_DIM * 2);
  __bf16* wt_f2 = (__bf16*)carve((size_t)Z_DIM * H_DIM * 2);
  __bf16* wt_p1 = (__bf16*)carve((size_t)H_DIM * (R_DIM + Z_DIM) * 2);
  __bf16* wt_p2 = (__bf16*)carve((size_t)D_DIM * H_DIM * 2);
  // fp32 activations
  float* zstate = (float*)carve((size_t)B_DIM * S_DIM * Z_DIM * 4);
  float* Ah     = (float*)carve((size_t)B_DIM * H_DIM * 4);
  float* Aph    = (float*)carve((size_t)B_DIM * H_DIM * 4);
  float* Dh     = (float*)carve((size_t)B_DIM * H_DIM * 4);
  float* Ph     = (float*)carve((size_t)B_DIM * H_DIM * 4);
  float* Bz     = (float*)carve((size_t)B_DIM * S_DIM * H_DIM * 4);
  float* Pz     = (float*)carve((size_t)B_DIM * S_DIM * H_DIM * 4);
  float* sigma  = (float*)carve((size_t)B_DIM * Z_DIM * 4);
  float* prior  = (float*)carve((size_t)B_DIM * S_DIM * Z_DIM * 4);
  float* poster = (float*)carve((size_t)B_DIM * S_DIM * Z_DIM * 4);
  float* pbuf   = (float*)carve((size_t)B_DIM * S_DIM * D_DIM * 4);
  float* accum  = (float*)carve(256);

  zero_accum_kernel<<<1, 32, 0, stream>>>(accum);

  auto tw = [&](const float* W, __bf16* Wt, int K, int N) {
    int total = K * N;
    transpose_bf16_kernel<<<(total + 255) / 256, 256, 0, stream>>>(W, Wt, K, N);
  };
  tw(drift_w1, wt_d1, R_DIM + Z_DIM, H_DIM);
  tw(drift_w2, wt_d2, H_DIM, Z_DIM);
  tw(diff_w1,  wt_f1, R_DIM, H_DIM);
  tw(diff_w2,  wt_f2, H_DIM, Z_DIM);
  tw(p_w1,     wt_p1, R_DIM + Z_DIM, H_DIM);
  tw(p_w2,     wt_p2, H_DIM, D_DIM);

  z0_kernel<<<B_DIM, CH_DIM, 0, stream>>>(cov, cov_w1, cov_b1, cov_w2, cov_b2, zstate);

  // plain GEMM (no compose, no relu-in, identity out)
  auto gemm0 = [&](const float* A0g, const __bf16* Wtg, int ldw, const float* biasg,
                   float* Cg, int Mg, int Ng, int Kg, int ldag) {
    dim3 grid(Ng / 64, Mg / 64);
    wmma_gemm_kernel<0, false, false><<<grid, 128, 0, stream>>>(
        A0g, nullptr, Wtg, ldw, biasg, Cg, Mg, Ng, Kg, ldag);
  };
  // relu-in + exp-out (diffusion)
  auto gemm_exp = [&](const float* A0g, const __bf16* Wtg, int ldw, const float* biasg,
                      float* Cg, int Mg, int Ng, int Kg, int ldag) {
    dim3 grid(Ng / 64, Mg / 64);
    wmma_gemm_kernel<2, false, true><<<grid, 128, 0, stream>>>(
        A0g, nullptr, Wtg, ldw, biasg, Cg, Mg, Ng, Kg, ldag);
  };
  // compose(A0 + bcast A1) + relu-in + 0.1*tanh-out (drifts)
  auto gemm_tanh = [&](const float* A0g, const float* A1g, const __bf16* Wtg, int ldw,
                       const float* biasg, float* Cg, int Mg, int Ng, int Kg, int ldag) {
    dim3 grid(Ng / 64, Mg / 64);
    wmma_gemm_kernel<3, true, true><<<grid, 128, 0, stream>>>(
        A0g, A1g, Wtg, ldw, biasg, Cg, Mg, Ng, Kg, ldag);
  };
  // compose(A0 + bcast A1) + relu-in, identity out (decoder layer 2)
  auto gemm_cmp = [&](const float* A0g, const float* A1g, const __bf16* Wtg, int ldw,
                      const float* biasg, float* Cg, int Mg, int Ng, int Kg, int ldag) {
    dim3 grid(Ng / 64, Mg / 64);
    wmma_gemm_kernel<0, true, true><<<grid, 128, 0, stream>>>(
        A0g, A1g, Wtg, ldw, biasg, Cg, Mg, Ng, Kg, ldag);
  };

  const int BS = B_DIM * S_DIM;
  const int LDW1 = R_DIM + Z_DIM;   // 768
  for (int t = 0; t < T_STEPS; ++t) {
    const float* h_t   = path_h    + (size_t)t * B_DIM * R_DIM;
    const float* hp_t  = path_hpos + (size_t)t * B_DIM * R_DIM;
    const float* X_t   = X     + (size_t)t * B_DIM * D_DIM;
    const float* M_t   = Mmask + (size_t)t * B_DIM * D_DIM;
    const float* eps_t = noise + (size_t)t * BS * Z_DIM;

    // per-batch hidden pre-activations (h part of W1, bias folded here)
    gemm0(h_t,  wt_d1, LDW1,  drift_b1, Ah,  B_DIM, H_DIM, R_DIM, R_DIM);
    gemm0(hp_t, wt_d1, LDW1,  drift_b1, Aph, B_DIM, H_DIM, R_DIM, R_DIM);
    gemm0(h_t,  wt_f1, R_DIM, diff_b1,  Dh,  B_DIM, H_DIM, R_DIM, R_DIM);
    gemm0(h_t,  wt_p1, LDW1,  p_b1,     Ph,  B_DIM, H_DIM, R_DIM, R_DIM);
    // z part of drift W1 (shared by prior & posterior)
    gemm0(zstate, wt_d1 + R_DIM, LDW1, nullptr, Bz, BS, H_DIM, Z_DIM, Z_DIM);
    // diffusion = exp(relu(Dh) @ W2 + b2)   [per batch, broadcast over S]
    gemm_exp(Dh, wt_f2, H_DIM, diff_b2, sigma, B_DIM, Z_DIM, H_DIM, H_DIM);
    // drifts = 0.1*tanh(relu(Ah/Aph broadcast + Bz) @ W2 + b2)  (A compose fused)
    gemm_tanh(Bz, Ah,  wt_d2, H_DIM, drift_b2, prior,  BS, Z_DIM, H_DIM, H_DIM);
    gemm_tanh(Bz, Aph, wt_d2, H_DIM, drift_b2, poster, BS, Z_DIM, H_DIM, H_DIM);
    // EM step + min-max normalize + KLD
    update_kernel<<<BS / 8, 256, 0, stream>>>(prior, poster, sigma, eps_t, zstate, accum);
    // decoder: Pz = z_new @ pW1_z ; p = relu(Ph + Pz) @ pW2 + b2
    gemm0(zstate, wt_p1 + R_DIM, LDW1, nullptr, Pz, BS, H_DIM, Z_DIM, Z_DIM);
    gemm_cmp(Pz, Ph, wt_p2, H_DIM, p_b2, pbuf, BS, D_DIM, H_DIM, H_DIM);
    // reconstruction NLL
    recon_kernel<<<B_DIM, D_DIM, 0, stream>>>(X_t, M_t, pbuf, accum);
  }

  finalize_kernel<<<(BS * Z_DIM + 255) / 256, 256, 0, stream>>>(zstate, accum, (float*)d_out);
}
